// HeterogeneousEncoder_43258910605921
// MI455X (gfx1250) — matrix operand order repaired and checked by reading, not compiled, not run on hardware
//
#include <hip/hip_runtime.h>

// ---------------------------------------------------------------------------
// HeterogeneousEncoder on MI455X (gfx1250): all GEMMs + attention through
// V_WMMA_F32_16X16X32_BF16 (fp32 accumulate), elementwise math in fp32.
// Weights pre-converted to bf16 and pre-transposed (N-major) so GEMM
// fragments load contiguously from global with no LDS staging.
// ---------------------------------------------------------------------------

typedef __attribute__((ext_vector_type(16))) __bf16 v16bf;
typedef __attribute__((ext_vector_type(8)))  __bf16 v8bf;
typedef __attribute__((ext_vector_type(8)))  float  v8f;

#define D_MODEL 256
#define S_FULL  800
#define NDRUG   400
#define BATCH   8
#define HEADS   8
#define DEPTH   32
#define DFF     512

__device__ __forceinline__ v16bf mk16(v8bf lo, v8bf hi) {
  v16bf r;
#pragma unroll
  for (int i = 0; i < 8; ++i) { r[i] = lo[i]; r[8 + i] = hi[i]; }
  return r;
}

__device__ __forceinline__ v8f wmma_bf16(v16bf a, v16bf b, v8f c) {
  return __builtin_amdgcn_wmma_f32_16x16x32_bf16(false, a, false, b, (short)0, c,
                                                 false, false);
}

__device__ __forceinline__ v8f vzero8() {
  v8f z = {0.f, 0.f, 0.f, 0.f, 0.f, 0.f, 0.f, 0.f};
  return z;
}

// ---------------------------------------------------------------------------
// f32 -> bf16 elementwise convert
// ---------------------------------------------------------------------------
__global__ void cvt_kernel(const float* __restrict__ in, __bf16* __restrict__ out, int n) {
  int i = blockIdx.x * blockDim.x + threadIdx.x;
  if (i < n) out[i] = (__bf16)in[i];
}

// ---------------------------------------------------------------------------
// weight prep: W[K,N] f32 (row-major) -> Wt[N,K] bf16 (transposed)
// ---------------------------------------------------------------------------
__global__ void wtr_kernel(const float* __restrict__ in, __bf16* __restrict__ out,
                           int K, int N) {
  int i = blockIdx.x * blockDim.x + threadIdx.x;
  if (i < K * N) {
    int k = i / N, n = i % N;
    out[(size_t)n * K + k] = (__bf16)in[i];
  }
}

// ---------------------------------------------------------------------------
// GEMM: C[M,N] = act(A[M,K](bf16,row-major) @ Wt[N,K](bf16,N-major) + bias)
// Block: 128 threads (4 waves). Block tile 128x64; wave tile 32x64.
// No LDS, no barriers: A- and B-fragments are contiguous 16B global loads
// with A and B both in k = 16*hi + s order. 8 WMMAs per wave per K-step.
// ACT: 0=none 1=relu 2=gelu(exact)
// ---------------------------------------------------------------------------
template <int ACT, bool WF32, bool WBF16>
__global__ __launch_bounds__(128) void gemm_kernel(
    const __bf16* __restrict__ A, const __bf16* __restrict__ Wt,
    const float* __restrict__ bias, float* __restrict__ Cf,
    __bf16* __restrict__ Cbf, int M, int N, int K) {
  const int tid = threadIdx.x;
  const int wave = tid >> 5;
  const int lane = tid & 31;
  const int l = lane & 15;
  const int hi = lane >> 4;
  const int m0 = blockIdx.x * 128 + wave * 32;
  const int n0 = blockIdx.y * 64;

  v8f acc[2][4];
#pragma unroll
  for (int i = 0; i < 2; ++i)
#pragma unroll
    for (int j = 0; j < 4; ++j) acc[i][j] = vzero8();

  const __bf16* arow0 = A + (size_t)(m0 + l) * K;
  const __bf16* arow1 = A + (size_t)(m0 + 16 + l) * K;
  const __bf16* brow[4];
#pragma unroll
  for (int j = 0; j < 4; ++j)
    brow[j] = Wt + (size_t)(n0 + 16 * j + l) * K + hi * 16;

  const int ka_off = hi ? 16 : 0;
  const int kb_off = hi ? 24 : 8;

  for (int k0 = 0; k0 < K; k0 += 32) {
    if (k0 + 32 < K) {  // prefetch next A slices (global_prefetch_b8)
      __builtin_prefetch(arow0 + k0 + 32, 0, 1);
      __builtin_prefetch(arow1 + k0 + 32, 0, 1);
    }
    v16bf a0 = mk16(*(const v8bf*)(arow0 + k0 + ka_off),
                    *(const v8bf*)(arow0 + k0 + kb_off));
    v16bf a1 = mk16(*(const v8bf*)(arow1 + k0 + ka_off),
                    *(const v8bf*)(arow1 + k0 + kb_off));
#pragma unroll
    for (int j = 0; j < 4; ++j) {
      v16bf bf = mk16(*(const v8bf*)(brow[j] + k0),
                      *(const v8bf*)(brow[j] + k0 + 8));
      acc[0][j] = wmma_bf16(a0, bf, acc[0][j]);
      acc[1][j] = wmma_bf16(a1, bf, acc[1][j]);
    }
  }

  // epilogue
#pragma unroll
  for (int j = 0; j < 4; ++j) {
    int col = n0 + j * 16 + l;
    float bv = bias ? bias[col] : 0.f;
#pragma unroll
    for (int i = 0; i < 2; ++i) {
#pragma unroll
      for (int r = 0; r < 8; ++r) {
        int row = m0 + i * 16 + hi * 8 + r;
        float v = acc[i][j][r] + bv;
        if (ACT == 1) v = fmaxf(v, 0.f);
        if (ACT == 2) v = 0.5f * v * (1.f + erff(v * 0.70710678118654752f));
        size_t idx = (size_t)row * N + col;
        if (WF32)  Cf[idx]  = v;
        if (WBF16) Cbf[idx] = (__bf16)v;
      }
    }
  }
}

// ---------------------------------------------------------------------------
// local-attention bias gather (never materialized)
// ---------------------------------------------------------------------------
__device__ __forceinline__ float bias_at(const float* __restrict__ dadj,
                                         const float* __restrict__ tadj,
                                         const float* __restrict__ inter,
                                         int b, int q, int k) {
  if (q < NDRUG) {
    if (k < NDRUG) return dadj[((size_t)b * NDRUG + q) * NDRUG + k];
    return inter[((size_t)b * NDRUG + q) * NDRUG + (k - NDRUG)];
  } else {
    if (k < NDRUG) return inter[((size_t)b * NDRUG + k) * NDRUG + (q - NDRUG)];
    return tadj[((size_t)b * NDRUG + (q - NDRUG)) * NDRUG + (k - NDRUG)];
  }
}

// ---------------------------------------------------------------------------
// Flash attention: block = (b, h, 32 q rows), 2 waves x 16 rows.
// depth=32 -> QK^T per 16x16 tile is ONE bf16 WMMA. Keys chunked by 32.
// ---------------------------------------------------------------------------
template <bool LOCAL>
__global__ __launch_bounds__(64) void attn_kernel(
    const __bf16* __restrict__ Q, const __bf16* __restrict__ Kb,
    const __bf16* __restrict__ V, const float* __restrict__ dadj,
    const float* __restrict__ tadj, const float* __restrict__ inter,
    __bf16* __restrict__ Out) {
  const int b = blockIdx.z, h = blockIdx.y;
  const int wave = threadIdx.x >> 5;
  const int lane = threadIdx.x & 31;
  const int l = lane & 15, hi = lane >> 4;
  const int q0 = blockIdx.x * 32 + wave * 16;

  __shared__ __align__(16) __bf16 Vt[32 * 32];       // [d][key], shared by waves
  __shared__ __align__(16) __bf16 Pl[2][16 * 32];    // per-wave P scratch

  // Q fragment: rows q0..q0+15, K-dim = depth 0..31
  const __bf16* qp = Q + ((size_t)b * S_FULL + q0 + l) * D_MODEL + h * DEPTH;
  v16bf qf = mk16(*(const v8bf*)(qp + (hi ? 16 : 0)),
                  *(const v8bf*)(qp + (hi ? 24 : 8)));

  float m[8], lsum[8];
#pragma unroll
  for (int r = 0; r < 8; ++r) { m[r] = -1e30f; lsum[r] = 0.f; }
  v8f acc0 = vzero8(), acc1 = vzero8();
  const float sc = 0.17677669529663687f;  // 1/sqrt(32)

  for (int k0 = 0; k0 < S_FULL; k0 += 32) {
    // K fragments straight from global (contiguous 32B per lane)
    const __bf16* kp0 = Kb + ((size_t)b * S_FULL + k0 + l) * D_MODEL + h * DEPTH + hi * 16;
    const __bf16* kp1 = Kb + ((size_t)b * S_FULL + k0 + 16 + l) * D_MODEL + h * DEPTH + hi * 16;
    v16bf kf0 = mk16(*(const v8bf*)kp0, *(const v8bf*)(kp0 + 8));
    v16bf kf1 = mk16(*(const v8bf*)kp1, *(const v8bf*)(kp1 + 8));
    v8f s0 = wmma_bf16(qf, kf0, vzero8());
    v8f s1 = wmma_bf16(qf, kf1, vzero8());

    // stage V chunk transposed: Vt[d][key]
    __syncthreads();
    {
      int key = threadIdx.x >> 1;
      int dbase = (threadIdx.x & 1) * 16;
      const __bf16* vp = V + ((size_t)b * S_FULL + k0 + key) * D_MODEL + h * DEPTH + dbase;
#pragma unroll
      for (int i = 0; i < 16; ++i) Vt[(dbase + i) * 32 + key] = vp[i];
    }
    __syncthreads();

    // online softmax update (rows mapped per C-fragment layout)
#pragma unroll
    for (int r = 0; r < 8; ++r) {
      int qr = q0 + hi * 8 + r;
      float v0 = s0[r] * sc, v1 = s1[r] * sc;
      if (LOCAL) {
        v0 += bias_at(dadj, tadj, inter, b, qr, k0 + l);
        v1 += bias_at(dadj, tadj, inter, b, qr, k0 + 16 + l);
      }
      float t = fmaxf(v0, v1);
      t = fmaxf(t, __shfl_xor(t, 1));
      t = fmaxf(t, __shfl_xor(t, 2));
      t = fmaxf(t, __shfl_xor(t, 4));
      t = fmaxf(t, __shfl_xor(t, 8));
      float mn = fmaxf(m[r], t);
      float sca = __expf(m[r] - mn);
      m[r] = mn;
      float p0 = __expf(v0 - mn), p1 = __expf(v1 - mn);
      float ps = p0 + p1;
      ps += __shfl_xor(ps, 1);
      ps += __shfl_xor(ps, 2);
      ps += __shfl_xor(ps, 4);
      ps += __shfl_xor(ps, 8);
      lsum[r] = lsum[r] * sca + ps;
      acc0[r] *= sca;
      acc1[r] *= sca;
      Pl[wave][(hi * 8 + r) * 32 + l]      = (__bf16)p0;
      Pl[wave][(hi * 8 + r) * 32 + 16 + l] = (__bf16)p1;
    }

    // P (C-layout) -> A-fragment through per-wave LDS (in-order DS, same wave)
    v16bf pf = mk16(*(const v8bf*)&Pl[wave][l * 32 + (hi ? 16 : 0)],
                    *(const v8bf*)&Pl[wave][l * 32 + (hi ? 24 : 8)]);
    // V B-fragments from Vt: contiguous rows
    v16bf vf0 = mk16(*(const v8bf*)&Vt[l * 32 + hi * 16],
                     *(const v8bf*)&Vt[l * 32 + hi * 16 + 8]);
    v16bf vf1 = mk16(*(const v8bf*)&Vt[(16 + l) * 32 + hi * 16],
                     *(const v8bf*)&Vt[(16 + l) * 32 + hi * 16 + 8]);
    acc0 = wmma_bf16(pf, vf0, acc0);
    acc1 = wmma_bf16(pf, vf1, acc1);
  }

#pragma unroll
  for (int r = 0; r < 8; ++r) {
    int qr = q0 + hi * 8 + r;
    float inv = 1.f / lsum[r];
    __bf16* op = Out + ((size_t)b * S_FULL + qr) * D_MODEL + h * DEPTH;
    op[l]      = (__bf16)(acc0[r] * inv);
    op[16 + l] = (__bf16)(acc1[r] * inv);
  }
}

// ---------------------------------------------------------------------------
// LayerNorm over D=256, one row per 256-thread block.
// mode 0: proj rows (b*400+i) -> X row b*800+seg+i, +pos[seg+i]
// mode 1: add+LN in residual stream (in-row == out-row)
// mode 2: final extract: in row b*800+i -> out row b*400+i
// ---------------------------------------------------------------------------
__global__ __launch_bounds__(256) void ln_kernel(
    const float* __restrict__ Yin, const float* __restrict__ Res,
    const float* __restrict__ g, const float* __restrict__ beta,
    const float* __restrict__ pos, float* __restrict__ OutF,
    __bf16* __restrict__ OutBf, int mode, int seg, int relu) {
  __shared__ float red[8];
  int rr = blockIdx.x, c = threadIdx.x;
  size_t inrow, outrow;
  int posrow = -1;
  if (mode == 0) {
    int b = rr / NDRUG, i = rr % NDRUG;
    inrow = rr;
    outrow = (size_t)b * S_FULL + seg + i;
    posrow = seg + i;
  } else if (mode == 1) {
    inrow = rr; outrow = rr;
  } else {
    int b = rr / NDRUG, i = rr % NDRUG;
    inrow = (size_t)b * S_FULL + i;
    outrow = rr;
  }
  float x = Yin[inrow * D_MODEL + c];
  if (relu) x = fmaxf(x, 0.f);
  if (Res) x += Res[inrow * D_MODEL + c];

  float s = x;
  s += __shfl_xor(s, 16); s += __shfl_xor(s, 8);
  s += __shfl_xor(s, 4);  s += __shfl_xor(s, 2); s += __shfl_xor(s, 1);
  if ((c & 31) == 0) red[c >> 5] = s;
  __syncthreads();
  float tot = 0.f;
#pragma unroll
  for (int i = 0; i < 8; ++i) tot += red[i];
  float mean = tot * (1.f / D_MODEL);
  float d = x - mean;
  float s2 = d * d;
  s2 += __shfl_xor(s2, 16); s2 += __shfl_xor(s2, 8);
  s2 += __shfl_xor(s2, 4);  s2 += __shfl_xor(s2, 2); s2 += __shfl_xor(s2, 1);
  __syncthreads();
  if ((c & 31) == 0) red[c >> 5] = s2;
  __syncthreads();
  float v = 0.f;
#pragma unroll
  for (int i = 0; i < 8; ++i) v += red[i];
  v *= (1.f / D_MODEL);
  float y = d * rsqrtf(v + 1e-5f) * g[c] + beta[c];
  if (posrow >= 0) y += pos[(size_t)posrow * D_MODEL + c];
  if (OutF)  OutF[outrow * D_MODEL + c]  = y;
  if (OutBf) OutBf[outrow * D_MODEL + c] = (__bf16)y;
}

// ---------------------------------------------------------------------------
// Host launch
// ---------------------------------------------------------------------------
extern "C" void kernel_launch(void* const* d_in, const int* in_sizes, int n_in,
                              void* d_out, int out_size, void* d_ws, size_t ws_size,
                              hipStream_t stream) {
  (void)in_sizes; (void)n_in; (void)out_size; (void)ws_size;
  const float* f_drug   = (const float*)d_in[0];  // (8,400,64)
  const float* f_target = (const float*)d_in[1];  // (8,400,128)
  const float* dadj     = (const float*)d_in[2];  // (8,400,400)
  const float* tadj     = (const float*)d_in[3];  // (8,400,400)
  const float* inter    = (const float*)d_in[4];  // (8,400,400)
  // params leaves (jax tree_flatten: dict keys sorted, lists in order)
  auto P = [&](int i) { return (const float*)d_in[5 + i]; };
  const float *dp_b = P(0), *dp_beta = P(1), *dp_g = P(2), *dp_w = P(3);
  const float *ex_b = P(4), *ex_beta = P(5), *ex_g = P(6), *ex_w = P(7);
  const float* pos = P(8 + 26 * 4);
  const float *tp_b = P(8 + 26 * 4 + 1), *tp_beta = P(8 + 26 * 4 + 2),
              *tp_g = P(8 + 26 * 4 + 3), *tp_w = P(8 + 26 * 4 + 4);

  // workspace carve
  char* w = (char*)d_ws;
  auto carve = [&](size_t bytes) {
    void* p = (void*)w;
    w += (bytes + 255) & ~(size_t)255;
    return p;
  };
  const size_t MROWS = (size_t)BATCH * S_FULL;  // 6400
  float*  Xf  = (float*)carve(MROWS * D_MODEL * 4);
  __bf16* Xbf = (__bf16*)carve(MROWS * D_MODEL * 2);
  __bf16* Qbf = (__bf16*)carve(MROWS * D_MODEL * 2);
  __bf16* Kbf = (__bf16*)carve(MROWS * D_MODEL * 2);
  __bf16* Vbf = (__bf16*)carve(MROWS * D_MODEL * 2);
  __bf16* Abf = (__bf16*)carve(MROWS * D_MODEL * 2);
  __bf16* Hbf = (__bf16*)carve(MROWS * DFF * 2);
  float*  Yf  = (float*)carve(MROWS * DFF * 4);
  __bf16* Fbf = (__bf16*)carve((size_t)BATCH * NDRUG * 128 * 2);
  // bf16 transposed weights
  __bf16* Wall = (__bf16*)carve((size_t)(64 * 256 + 128 * 256 + 256 * 256 +
                                         4 * (8 * 256 * 256 + 2 * 256 * 512)) * 2);
  size_t woff = 0;
  auto wslot = [&](size_t elems) { __bf16* p = Wall + woff; woff += elems; return p; };
  __bf16* dpWt = wslot(64 * 256);
  __bf16* tpWt = wslot(128 * 256);
  __bf16* exWt = wslot(256 * 256);
  __bf16 *lqWt[4], *lkWt[4], *lvWt[4], *loWt[4];
  __bf16 *gqWt[4], *gkWt[4], *gvWt[4], *goWt[4];
  __bf16 *f1Wt[4], *f2Wt[4];
  for (int lyr = 0; lyr < 4; ++lyr) {
    lqWt[lyr] = wslot(256 * 256); lkWt[lyr] = wslot(256 * 256);
    lvWt[lyr] = wslot(256 * 256); loWt[lyr] = wslot(256 * 256);
    gqWt[lyr] = wslot(256 * 256); gkWt[lyr] = wslot(256 * 256);
    gvWt[lyr] = wslot(256 * 256); goWt[lyr] = wslot(256 * 256);
    f1Wt[lyr] = wslot(256 * 512); f2Wt[lyr] = wslot(512 * 256);
  }

  auto wtr = [&](const float* src, __bf16* dst, int K, int N) {
    wtr_kernel<<<(K * N + 255) / 256, 256, 0, stream>>>(src, dst, K, N);
  };

  // ---- weight prep (once per launch) ----
  wtr(dp_w, dpWt, 64, 256);
  wtr(tp_w, tpWt, 128, 256);
  wtr(ex_w, exWt, 256, 256);
  for (int lyr = 0; lyr < 4; ++lyr) {
    int L = 8 + 26 * lyr;
    wtr(P(L + 23), lqWt[lyr], 256, 256);  // l_q.w
    wtr(P(L + 19), lkWt[lyr], 256, 256);  // l_k.w
    wtr(P(L + 25), lvWt[lyr], 256, 256);  // l_v.w
    wtr(P(L + 21), loWt[lyr], 256, 256);  // l_o.w
    wtr(P(L + 15), gqWt[lyr], 256, 256);  // g_q.w
    wtr(P(L + 11), gkWt[lyr], 256, 256);  // g_k.w
    wtr(P(L + 17), gvWt[lyr], 256, 256);  // g_v.w
    wtr(P(L + 13), goWt[lyr], 256, 256);  // g_o.w
    wtr(P(L + 1),  f1Wt[lyr], 256, 512);  // ffn1.w
    wtr(P(L + 3),  f2Wt[lyr], 512, 256);  // ffn2.w
  }

  const dim3 gN(50, 4);   // M=6400, N=256  (128x64 block tiles)
  const dim3 gH(50, 8);   // M=6400, N=512
  const dim3 gP(25, 4);   // M=3200, N=256
  const dim3 gA(25, HEADS, BATCH);

  // ---- input projections ----
  cvt_kernel<<<(BATCH * NDRUG * 64 + 255) / 256, 256, 0, stream>>>(f_drug, Fbf, BATCH * NDRUG * 64);
  gemm_kernel<0, true, false><<<gP, 128, 0, stream>>>(Fbf, dpWt, dp_b, Yf, nullptr, 3200, 256, 64);
  ln_kernel<<<3200, 256, 0, stream>>>(Yf, nullptr, dp_g, dp_beta, pos, Xf, Xbf, 0, 0, 1);

  cvt_kernel<<<(BATCH * NDRUG * 128 + 255) / 256, 256, 0, stream>>>(f_target, Fbf, BATCH * NDRUG * 128);
  gemm_kernel<0, true, false><<<gP, 128, 0, stream>>>(Fbf, tpWt, tp_b, Yf, nullptr, 3200, 256, 128);
  ln_kernel<<<3200, 256, 0, stream>>>(Yf, nullptr, tp_g, tp_beta, pos, Xf, Xbf, 0, 400, 1);

  // ---- encoder layers ----
  for (int lyr = 0; lyr < 4; ++lyr) {
    int L = 8 + 26 * lyr;
    const float *f1b = P(L + 0), *f2b = P(L + 2);
    const float *ln1b = P(L + 4), *ln1g = P(L + 5), *ln2b = P(L + 6), *ln2g = P(L + 7);
    const float *ln3b = P(L + 8), *ln3g = P(L + 9);
    const float *gkb = P(L + 10), *gob = P(L + 12), *gqb = P(L + 14), *gvb = P(L + 16);
    const float *lkb = P(L + 18), *lob = P(L + 20), *lqb = P(L + 22), *lvb = P(L + 24);

    // local MHA
    gemm_kernel<0, false, true><<<gN, 128, 0, stream>>>(Xbf, lqWt[lyr], lqb, nullptr, Qbf, 6400, 256, 256);
    gemm_kernel<0, false, true><<<gN, 128, 0, stream>>>(Xbf, lkWt[lyr], lkb, nullptr, Kbf, 6400, 256, 256);
    gemm_kernel<0, false, true><<<gN, 128, 0, stream>>>(Xbf, lvWt[lyr], lvb, nullptr, Vbf, 6400, 256, 256);
    attn_kernel<true><<<gA, 64, 0, stream>>>(Qbf, Kbf, Vbf, dadj, tadj, inter, Abf);
    gemm_kernel<0, true, false><<<gN, 128, 0, stream>>>(Abf, loWt[lyr], lob, Yf, nullptr, 6400, 256, 256);
    ln_kernel<<<6400, 256, 0, stream>>>(Yf, Xf, ln1g, ln1b, nullptr, Xf, Xbf, 1, 0, 0);

    // global MHA
    gemm_kernel<0, false, true><<<gN, 128, 0, stream>>>(Xbf, gqWt[lyr], gqb, nullptr, Qbf, 6400, 256, 256);
    gemm_kernel<0, false, true><<<gN, 128, 0, stream>>>(Xbf, gkWt[lyr], gkb, nullptr, Kbf, 6400, 256, 256);
    gemm_kernel<0, false, true><<<gN, 128, 0, stream>>>(Xbf, gvWt[lyr], gvb, nullptr, Vbf, 6400, 256, 256);
    attn_kernel<false><<<gA, 64, 0, stream>>>(Qbf, Kbf, Vbf, nullptr, nullptr, nullptr, Abf);
    gemm_kernel<0, true, false><<<gN, 128, 0, stream>>>(Abf, goWt[lyr], gob, Yf, nullptr, 6400, 256, 256);
    ln_kernel<<<6400, 256, 0, stream>>>(Yf, Xf, ln2g, ln2b, nullptr, Xf, Xbf, 1, 0, 0);

    // FFN
    gemm_kernel<2, false, true><<<gH, 128, 0, stream>>>(Xbf, f1Wt[lyr], f1b, nullptr, Hbf, 6400, 512, 256);
    gemm_kernel<0, true, false><<<gN, 128, 0, stream>>>(Hbf, f2Wt[lyr], f2b, Yf, nullptr, 6400, 256, 512);
    ln_kernel<<<6400, 256, 0, stream>>>(Yf, Xf, ln3g, ln3b, nullptr, Xf, Xbf, 1, 0, 0);
  }

  // ---- extract head (compute all rows, emit drug rows) ----
  gemm_kernel<0, true, false><<<gN, 128, 0, stream>>>(Xbf, exWt, ex_b, Yf, nullptr, 6400, 256, 256);
  ln_kernel<<<3200, 256, 0, stream>>>(Yf, nullptr, ex_g, ex_beta, nullptr,
                                      (float*)d_out, nullptr, 2, 0, 1);
}
